// Neighbor2PointAttention_17145509446486
// MI455X (gfx1250) — matrix operand order repaired and checked by reading, not compiled, not run on hardware
//
#include <hip/hip_runtime.h>
#include <hip/hip_bf16.h>

// ---------------------------------------------------------------------------
// Problem constants (match reference)
// ---------------------------------------------------------------------------
#define BB 8
#define CC 64
#define NN 2048
#define KNB 32          // neighbors
#define HEADS 8
#define DHEAD 8
#define CMID 128
#define EPS 1e-5f
#define NEG_SLOPE 0.2f

typedef __attribute__((ext_vector_type(16))) __bf16 bf16x16;
typedef __attribute__((ext_vector_type(8)))  float  floatx8;
typedef __attribute__((ext_vector_type(4)))  unsigned int uint32x4;
typedef __attribute__((ext_vector_type(8)))  int          int32x8;
typedef __attribute__((ext_vector_type(4)))  int          int32x4;

// ---------------------------------------------------------------------------
// WMMA helpers (V_WMMA_F32_16X16X32_BF16, probe-confirmed builtin)
// ---------------------------------------------------------------------------
__device__ __forceinline__ floatx8 wmma_bf16(bf16x16 a, bf16x16 b, floatx8 c) {
  return __builtin_amdgcn_wmma_f32_16x16x32_bf16(
      /*neg_a=*/false, a, /*neg_b=*/false, b,
      /*c_mod=*/(short)0, c, /*reuse_a=*/false, /*reuse_b=*/false);
}

// A fragment (16x32, MxK) from row-major bf16 matrix.
// Contiguous k order (same as load_b_T): lane L holds row M=L&15; element e -> K = k0 + (L>=16?16:0) + e
// -> 16 contiguous bf16 per lane.
__device__ __forceinline__ bf16x16 load_a_bf16(const __bf16* mat, int ld,
                                               int m0, int k0, int lane) {
  int m   = m0 + (lane & 15);
  int khi = (lane >> 4) ? 16 : 0;
  const __bf16* base = mat + (size_t)m * ld + k0 + khi;
  bf16x16 a;
#pragma unroll
  for (int e = 0; e < 8; ++e) a[e] = base[e];
#pragma unroll
  for (int e = 0; e < 8; ++e) a[8 + e] = base[8 + e];
  return a;
}

// B fragment (32x16, KxN). Source matT[point][feature] (row = N-col of B, col = K).
// Lane L holds column n=L&15; elements e=0..15 are K = k0+(L/16)*16+e
// -> 16 contiguous bf16 = one 32B vector load.
__device__ __forceinline__ bf16x16 load_b_T(const __bf16* matT, int ld, int k0,
                                            int n0, int lane) {
  int n     = n0 + (lane & 15);
  int kbase = k0 + (lane >> 4) * 16;
  return *(const bf16x16*)(matT + (size_t)n * ld + kbase);
}

// ---------------------------------------------------------------------------
// TDM: async 2-D tile load (tensor = NN x CC bf16 rows, tile = rows x CC)
// D# layout per CDNA5 ISA §8.3/§8.4. Groups 2/3 zero (<=2D tensor).
// ---------------------------------------------------------------------------
__device__ __forceinline__ void tdm_load_tile_2d(unsigned lds_addr,
                                                 unsigned long long gaddr,
                                                 int tile_rows) {
  uint32x4 g0;
  g0[0] = 1u;                                       // count=1 user descriptor
  g0[1] = lds_addr;                                 // LDS byte address
  g0[2] = (unsigned)(gaddr & 0xffffffffull);        // global_addr[31:0]
  g0[3] = (unsigned)((gaddr >> 32) & 0x1ffffffull)  // global_addr[56:32]
          | (2u << 30);                             // type=2 ("image")
  int32x8 g1;
  g1[0] = (1 << 16);                                // data_size=1 (2 bytes/elt)
  g1[1] = (CC & 0xffff) << 16;                      // tensor_dim0[15:0]
  g1[2] = (CC >> 16) | ((NN & 0xffff) << 16);       // tdim0[31:16] | tdim1[15:0]
  g1[3] = (NN >> 16) | (CC << 16);                  // tdim1[31:16] | tile_dim0
  g1[4] = tile_rows & 0xffff;                       // tile_dim1 (tile_dim2=0)
  g1[5] = CC;                                       // tensor_dim0_stride[31:0]
  g1[6] = 0;                                        // stride0 hi | stride1 lo
  g1[7] = 0;                                        // stride1 hi
  int32x4 z4 = {0, 0, 0, 0};
#if defined(__clang_major__) && (__clang_major__ >= 23)
  int32x8 z8 = {0, 0, 0, 0, 0, 0, 0, 0};
  __builtin_amdgcn_tensor_load_to_lds(g0, g1, z4, z4, z8, 0);
#else
  __builtin_amdgcn_tensor_load_to_lds(g0, g1, z4, z4, 0);
#endif
}

// ---------------------------------------------------------------------------
// Kernel 0a: transpose x -> bf16 (B,N,C) + per-point squared norms
// ---------------------------------------------------------------------------
__global__ void prep_kernel(const float* __restrict__ x, __bf16* __restrict__ xT,
                            float* __restrict__ xx) {
  int i = blockIdx.x * blockDim.x + threadIdx.x;  // b*N + n
  if (i >= BB * NN) return;
  int b = i / NN, n = i - b * NN;
  const float* xb = x + (size_t)b * CC * NN + n;
  __bf16* xo = xT + (size_t)i * CC;
  float s = 0.f;
#pragma unroll
  for (int c = 0; c < CC; ++c) {
    float v = xb[(size_t)c * NN];
    s += v * v;
    xo[c] = (__bf16)v;
  }
  xx[i] = s;
}

// ---------------------------------------------------------------------------
// Kernel 0b: one-shot weight conversion f32 -> bf16 (keeps cvt out of GEMMs)
// ---------------------------------------------------------------------------
__global__ void wconv_kernel(const float* __restrict__ Wq,
                             const float* __restrict__ Wk,
                             const float* __restrict__ Wv,
                             const float* __restrict__ W1,
                             const float* __restrict__ W2,
                             __bf16* __restrict__ wq, __bf16* __restrict__ wk,
                             __bf16* __restrict__ wv, __bf16* __restrict__ w1,
                             __bf16* __restrict__ w2) {
  int i = blockIdx.x * blockDim.x + threadIdx.x;
  if (i < CC * CC) {
    wq[i] = (__bf16)Wq[i];
    wk[i] = (__bf16)Wk[i];
    wv[i] = (__bf16)Wv[i];
  }
  if (i < CMID * CC) {
    w1[i] = (__bf16)W1[i];
    w2[i] = (__bf16)W2[i];
  }
}

// ---------------------------------------------------------------------------
// Kernel 1: Q/K/V projections. One wave per (batch, 16-point tile).
// Out stored point-major (B,N,64) for cheap neighbor gathers later.
// ---------------------------------------------------------------------------
__global__ void proj_kernel(const __bf16* __restrict__ xT,
                            const __bf16* __restrict__ Wq,
                            const __bf16* __restrict__ Wk,
                            const __bf16* __restrict__ Wv,
                            float* __restrict__ Q, float* __restrict__ Kx,
                            float* __restrict__ Vx) {
  int b  = blockIdx.x / (NN / 16);
  int n0 = (blockIdx.x % (NN / 16)) * 16;
  int lane = threadIdx.x;
  const __bf16* xb = xT + (size_t)b * NN * CC;
  bf16x16 b0 = load_b_T(xb, CC, 0, n0, lane);
  bf16x16 b1 = load_b_T(xb, CC, 32, n0, lane);
  const __bf16* Ws[3]  = {Wq, Wk, Wv};
  float*        Out[3] = {Q, Kx, Vx};
#pragma unroll
  for (int w = 0; w < 3; ++w) {
#pragma unroll
    for (int mt = 0; mt < 4; ++mt) {
      floatx8 acc = {};
      acc = wmma_bf16(load_a_bf16(Ws[w], CC, mt * 16, 0, lane), b0, acc);
      acc = wmma_bf16(load_a_bf16(Ws[w], CC, mt * 16, 32, lane), b1, acc);
      float* ob = Out[w] + ((size_t)b * NN + n0 + (lane & 15)) * CC + mt * 16 +
                  (lane >> 4) * 8;
#pragma unroll
      for (int r = 0; r < 8; ++r) ob[r] = acc[r];
    }
  }
}

// ---------------------------------------------------------------------------
// Kernel 2: kNN. One wave per (batch, 16-row tile). Column tiles are staged
// into LDS by the Tensor Data Mover (double-buffered, s_wait_tensorcnt),
// Gram tiles via WMMA, per-row top-32 kept in registers (lanes 0..15).
// ---------------------------------------------------------------------------
__global__ void knn_kernel(const __bf16* __restrict__ xT,
                           const float* __restrict__ xx,
                           int* __restrict__ idxout) {
  __shared__ __bf16 tile[2][16 * CC];  // 2 x 2KB column-tile staging
  __shared__ float s[16][17];
  int b  = blockIdx.x / (NN / 16);
  int n0 = (blockIdx.x % (NN / 16)) * 16;
  int lane = threadIdx.x;
  const __bf16* xb  = xT + (size_t)b * NN * CC;
  const float*  xxb = xx + (size_t)b * NN;

  // LDS offsets: low 32 bits of the flat shared address ARE the LDS offset
  // (ISA: LDS aperture occupies addr[63:32]).
  unsigned lds_buf[2];
  lds_buf[0] = (unsigned)(unsigned long long)(uintptr_t)&tile[0][0];
  lds_buf[1] = (unsigned)(unsigned long long)(uintptr_t)&tile[1][0];
  unsigned long long gbase = (unsigned long long)(uintptr_t)xb;

  bf16x16 a0 = load_a_bf16(xb, CC, n0, 0, lane);
  bf16x16 a1 = load_a_bf16(xb, CC, n0, 32, lane);

  float dist[KNB];
  int   ind[KNB];
#pragma unroll
  for (int j = 0; j < KNB; ++j) { dist[j] = -3.4e38f; ind[j] = 0; }
  float rxx = (lane < 16) ? xxb[n0 + lane] : 0.f;

  // Prime the pipeline: tile 0 -> buffer 0.
  tdm_load_tile_2d(lds_buf[0], gbase, 16);

  for (int m0 = 0; m0 < NN; m0 += 16) {
    int cur = (m0 >> 4) & 1;
    if (m0 + 16 < NN) {
      // Issue next tile into the other buffer, then wait for current tile.
      tdm_load_tile_2d(lds_buf[cur ^ 1], gbase + (size_t)(m0 + 16) * CC * 2, 16);
      __builtin_amdgcn_s_wait_tensorcnt(1);
    } else {
      __builtin_amdgcn_s_wait_tensorcnt(0);
    }
    const __bf16* tb = &tile[cur][0];
    bf16x16 b0 = load_b_T(tb, CC, 0, 0, lane);   // ds_load from staged tile
    bf16x16 b1 = load_b_T(tb, CC, 32, 0, lane);
    floatx8 acc = {};
    acc = wmma_bf16(a0, b0, acc);
    acc = wmma_bf16(a1, b1, acc);
    float cxx = xxb[m0 + (lane & 15)];
    int hi = (lane >> 4) * 8;
#pragma unroll
    for (int r = 0; r < 8; ++r)
      s[hi + r][lane & 15] = 2.f * acc[r] - cxx;  // neg_sq_dist + rxx (const/row)
    __syncthreads();
    if (lane < 16) {
#pragma unroll
      for (int cc = 0; cc < 16; ++cc) {
        float d = s[lane][cc] - rxx;
        if (d > dist[KNB - 1]) {
          float cd = d;
          int   ci = m0 + cc;
#pragma unroll
          for (int j = 0; j < KNB; ++j) {
            if (cd > dist[j]) {
              float td = dist[j]; int ti = ind[j];
              dist[j] = cd; ind[j] = ci;
              cd = td; ci = ti;
            }
          }
        }
      }
    }
    __syncthreads();
  }
  if (lane < 16) {
    int* op = idxout + ((size_t)b * NN + n0 + lane) * KNB;
#pragma unroll
    for (int j = 0; j < KNB; ++j) op[j] = ind[j];
  }
}

// ---------------------------------------------------------------------------
// Kernel 3: attention + residual. One wave per point; lane j = neighbor j.
// k_j = Kx[idx_j]-Kx[n], v_j likewise (linearity of the 1x1 conv).
// ---------------------------------------------------------------------------
__global__ void attn_kernel(const float* __restrict__ x,
                            const float* __restrict__ Qx,
                            const float* __restrict__ Kx,
                            const float* __restrict__ Vx,
                            const int* __restrict__ idx,
                            float* __restrict__ y1) {
  int wave = threadIdx.x >> 5;
  int lane = threadIdx.x & 31;
  int p = blockIdx.x * (blockDim.x >> 5) + wave;  // b*N + n
  if (p >= BB * NN) return;
  int b = p / NN, n = p - b * NN;
  int nb = idx[(size_t)p * KNB + lane];
  const float* qr = Qx + (size_t)p * CC;
  const float* kc = Kx + (size_t)p * CC;
  const float* vc = Vx + (size_t)p * CC;
  const float* kn = Kx + ((size_t)b * NN + nb) * CC;
  const float* vn = Vx + ((size_t)b * NN + nb) * CC;
  __builtin_prefetch(kn, 0, 3);
  __builtin_prefetch(vn, 0, 3);
  const float inv_scale = 0.35355339059327373f;  // 1/sqrt(8)

#pragma unroll
  for (int h = 0; h < HEADS; ++h) {
    float e = 0.f;
#pragma unroll
    for (int t = 0; t < DHEAD; ++t)
      e += qr[h * 8 + t] * (kn[h * 8 + t] - kc[h * 8 + t]);
    e *= inv_scale;
    float m = e;
#pragma unroll
    for (int sft = 16; sft > 0; sft >>= 1) m = fmaxf(m, __shfl_xor(m, sft, 32));
    float ex = __expf(e - m);
    float ssum = ex;
#pragma unroll
    for (int sft = 16; sft > 0; sft >>= 1) ssum += __shfl_xor(ssum, sft, 32);
    float attn = ex / ssum;
    float o[DHEAD];
#pragma unroll
    for (int t = 0; t < DHEAD; ++t) {
      float w = attn * (vn[h * 8 + t] - vc[h * 8 + t]);
#pragma unroll
      for (int sft = 16; sft > 0; sft >>= 1) w += __shfl_xor(w, sft, 32);
      o[t] = w;
    }
    if (lane == 0) {
#pragma unroll
      for (int t = 0; t < DHEAD; ++t) {
        size_t oi = ((size_t)b * CC + h * 8 + t) * NN + n;
        y1[oi] = x[oi] + o[t];
      }
    }
  }
}

// ---------------------------------------------------------------------------
// BatchNorm: stats (one block per channel) then apply (+ optional bf16^T copy)
// ---------------------------------------------------------------------------
__global__ void bn_stats_kernel(const float* __restrict__ y,
                                const float* __restrict__ gamma,
                                const float* __restrict__ beta,
                                float* __restrict__ scale,
                                float* __restrict__ shift) {
  __shared__ float ssum[256], ssq[256];
  int c = blockIdx.x, tid = threadIdx.x;
  float s = 0.f, sq = 0.f;
  for (int i = tid; i < BB * NN; i += 256) {
    int b = i / NN, n = i - b * NN;
    float v = y[((size_t)b * CC + c) * NN + n];
    s += v; sq += v * v;
  }
  ssum[tid] = s; ssq[tid] = sq;
  __syncthreads();
  for (int st = 128; st > 0; st >>= 1) {
    if (tid < st) { ssum[tid] += ssum[tid + st]; ssq[tid] += ssq[tid + st]; }
    __syncthreads();
  }
  if (tid == 0) {
    float mean = ssum[0] / (float)(BB * NN);
    float var  = ssq[0] / (float)(BB * NN) - mean * mean;
    float sc   = gamma[c] * rsqrtf(var + EPS);
    scale[c] = sc;
    shift[c] = beta[c] - mean * sc;
  }
}

__global__ void bn_apply_kernel(const float* __restrict__ y,
                                const float* __restrict__ scale,
                                const float* __restrict__ shift,
                                float* __restrict__ xout,
                                __bf16* __restrict__ xTout) {
  int i = blockIdx.x * 256 + threadIdx.x;
  if (i >= BB * CC * NN) return;
  int n = i & (NN - 1);
  int c = (i >> 11) & (CC - 1);
  int b = i >> 17;  // 64*2048 = 2^17
  float v = y[i] * scale[c] + shift[c];
  xout[i] = v;
  if (xTout) xTout[((size_t)b * NN + n) * CC + c] = (__bf16)v;
}

// ---------------------------------------------------------------------------
// FFN GEMM1: h = leaky_relu(W1 (128x64) * x1), output bf16 point-major (B,N,128)
// ---------------------------------------------------------------------------
__global__ void ffn1_kernel(const __bf16* __restrict__ x1T,
                            const __bf16* __restrict__ W1,
                            __bf16* __restrict__ hT) {
  int b  = blockIdx.x / (NN / 16);
  int n0 = (blockIdx.x % (NN / 16)) * 16;
  int lane = threadIdx.x;
  const __bf16* xb = x1T + (size_t)b * NN * CC;
  bf16x16 b0 = load_b_T(xb, CC, 0, n0, lane);
  bf16x16 b1 = load_b_T(xb, CC, 32, n0, lane);
#pragma unroll
  for (int mt = 0; mt < 8; ++mt) {
    floatx8 acc = {};
    acc = wmma_bf16(load_a_bf16(W1, CC, mt * 16, 0, lane), b0, acc);
    acc = wmma_bf16(load_a_bf16(W1, CC, mt * 16, 32, lane), b1, acc);
    __bf16* ob = hT + ((size_t)b * NN + n0 + (lane & 15)) * CMID + mt * 16 +
                 (lane >> 4) * 8;
#pragma unroll
    for (int r = 0; r < 8; ++r) {
      float v = acc[r];
      v = v > 0.f ? v : NEG_SLOPE * v;
      ob[r] = (__bf16)v;
    }
  }
}

// ---------------------------------------------------------------------------
// FFN GEMM2: y2 = x1 + W2 (64x128) * h
// ---------------------------------------------------------------------------
__global__ void ffn2_kernel(const __bf16* __restrict__ hT,
                            const __bf16* __restrict__ W2,
                            const float* __restrict__ x1,
                            float* __restrict__ y2) {
  int b  = blockIdx.x / (NN / 16);
  int n0 = (blockIdx.x % (NN / 16)) * 16;
  int lane = threadIdx.x;
  const __bf16* hb = hT + (size_t)b * NN * CMID;
  bf16x16 bf[4];
#pragma unroll
  for (int kk = 0; kk < 4; ++kk) bf[kk] = load_b_T(hb, CMID, kk * 32, n0, lane);
#pragma unroll
  for (int mt = 0; mt < 4; ++mt) {
    floatx8 acc = {};
#pragma unroll
    for (int kk = 0; kk < 4; ++kk)
      acc = wmma_bf16(load_a_bf16(W2, CMID, mt * 16, kk * 32, lane), bf[kk], acc);
    int n = n0 + (lane & 15);
#pragma unroll
    for (int r = 0; r < 8; ++r) {
      int m = mt * 16 + (lane >> 4) * 8 + r;
      size_t oi = ((size_t)b * CC + m) * NN + n;
      y2[oi] = x1[oi] + acc[r];
    }
  }
}

// ---------------------------------------------------------------------------
// Launch
// ---------------------------------------------------------------------------
extern "C" void kernel_launch(void* const* d_in, const int* in_sizes, int n_in,
                              void* d_out, int out_size, void* d_ws,
                              size_t ws_size, hipStream_t stream) {
  const float* x  = (const float*)d_in[0];
  const float* Wq = (const float*)d_in[1];
  const float* Wk = (const float*)d_in[2];
  const float* Wv = (const float*)d_in[3];
  const float* W1 = (const float*)d_in[4];
  const float* W2 = (const float*)d_in[5];
  const float* g1 = (const float*)d_in[6];
  const float* b1 = (const float*)d_in[7];
  const float* g2 = (const float*)d_in[8];
  const float* b2 = (const float*)d_in[9];
  float* out = (float*)d_out;

  char* ws = (char*)d_ws;
  size_t off = 0;
  auto take = [&](size_t bytes) {
    char* p = ws + off;
    off += (bytes + 255) & ~(size_t)255;
    return p;
  };
  __bf16* xT   = (__bf16*)take((size_t)BB * NN * CC * 2);
  float*  xx   = (float*) take((size_t)BB * NN * 4);
  float*  Qx   = (float*) take((size_t)BB * NN * CC * 4);
  float*  Kx   = (float*) take((size_t)BB * NN * CC * 4);
  float*  Vx   = (float*) take((size_t)BB * NN * CC * 4);
  int*    idx  = (int*)   take((size_t)BB * NN * KNB * 4);
  float*  y1   = (float*) take((size_t)BB * CC * NN * 4);
  float*  x1   = (float*) take((size_t)BB * CC * NN * 4);
  __bf16* x1T  = (__bf16*)take((size_t)BB * NN * CC * 2);
  __bf16* hT   = (__bf16*)take((size_t)BB * NN * CMID * 2);
  float*  y2   = (float*) take((size_t)BB * CC * NN * 4);
  __bf16* wqb  = (__bf16*)take((size_t)CC * CC * 2);
  __bf16* wkb  = (__bf16*)take((size_t)CC * CC * 2);
  __bf16* wvb  = (__bf16*)take((size_t)CC * CC * 2);
  __bf16* w1b  = (__bf16*)take((size_t)CMID * CC * 2);
  __bf16* w2b  = (__bf16*)take((size_t)CC * CMID * 2);
  float*  sc1  = (float*) take(CC * 4);
  float*  sh1  = (float*) take(CC * 4);
  float*  sc2  = (float*) take(CC * 4);
  float*  sh2  = (float*) take(CC * 4);

  const int tiles = BB * (NN / 16);  // 1024 single-wave blocks

  prep_kernel<<<(BB * NN + 255) / 256, 256, 0, stream>>>(x, xT, xx);
  wconv_kernel<<<(CMID * CC + 255) / 256, 256, 0, stream>>>(
      Wq, Wk, Wv, W1, W2, wqb, wkb, wvb, w1b, w2b);
  proj_kernel<<<tiles, 32, 0, stream>>>(xT, wqb, wkb, wvb, Qx, Kx, Vx);
  knn_kernel<<<tiles, 32, 0, stream>>>(xT, xx, idx);
  attn_kernel<<<BB * NN / 8, 256, 0, stream>>>(x, Qx, Kx, Vx, idx, y1);
  bn_stats_kernel<<<CC, 256, 0, stream>>>(y1, g1, b1, sc1, sh1);
  bn_apply_kernel<<<(BB * CC * NN + 255) / 256, 256, 0, stream>>>(y1, sc1, sh1,
                                                                  x1, x1T);
  ffn1_kernel<<<tiles, 32, 0, stream>>>(x1T, w1b, hT);
  ffn2_kernel<<<tiles, 32, 0, stream>>>(hT, w2b, x1, y2);
  bn_stats_kernel<<<CC, 256, 0, stream>>>(y2, g2, b2, sc2, sh2);
  bn_apply_kernel<<<(BB * CC * NN + 255) / 256, 256, 0, stream>>>(
      y2, sc2, sh2, out, (__bf16*)nullptr);
}
